// HeteroGNNLinkPredictor_84404697301637
// MI455X (gfx1250) — compile-verified, matrix operand order checked by reading, not run on hardware
//
#include <hip/hip_runtime.h>

typedef __attribute__((ext_vector_type(16))) __bf16 v16bf;
typedef __attribute__((ext_vector_type(8)))  float  v8f;

#define DFEAT 128

__global__ void fill_zero(float* __restrict__ p, long n) {
  long i = (long)blockIdx.x * blockDim.x + threadIdx.x;
  if (i < n) p[i] = 0.0f;
}

__global__ void count_edges(const int* __restrict__ dst, float* __restrict__ cnt, int E) {
  int e = blockIdx.x * blockDim.x + threadIdx.x;
  if (e < E) unsafeAtomicAdd(&cnt[dst[e]], 1.0f);
}

__global__ void invert_counts(float* __restrict__ cnt, int n) {
  int i = blockIdx.x * blockDim.x + threadIdx.x;
  if (i < n) cnt[i] = 1.0f / fmaxf(cnt[i], 1.0f);
}

// Transpose 128x128 f32 row-major W[k][n] into bf16 N-major Wt[n][k].
__global__ void transpose_w_bf16(const float* __restrict__ W, __bf16* __restrict__ Wt) {
  int i = blockIdx.x * blockDim.x + threadIdx.x;   // 16384 threads
  int k = i >> 7, n = i & 127;
  Wt[(long)n * DFEAT + k] = (__bf16)W[i];          // coalesced read, strided write (L2)
}

// One wave per edge; each lane moves a float4 slice of the 128-wide feature row.
__global__ void scatter_add(const float* __restrict__ x, const int* __restrict__ src,
                            const int* __restrict__ dst, float* __restrict__ sum, int E) {
  long t = (long)blockIdx.x * blockDim.x + threadIdx.x;
  int e = (int)(t >> 5);
  if (e >= E) return;
  int c = ((int)t & 31) * 4;
  const float4 v = *(const float4*)(x + (long)src[e] * DFEAT + c);
  float* d = sum + (long)dst[e] * DFEAT + c;
  unsafeAtomicAdd(d + 0, v.x);
  unsafeAtomicAdd(d + 1, v.y);
  unsafeAtomicAdd(d + 2, v.z);
  unsafeAtomicAdd(d + 3, v.w);
}

// out[M x 128] = act( diag(inv)*(Asum @ Wl) + H @ Wr + b )
// WlT/WrT are bf16, N-major (transposed). Block = 256 threads = 8 wave32s;
// block covers rows [16*bx, 16*bx+16), wave w covers the 16-col tile at 16*w.
// 8 x v_wmma_f32_16x16x32_bf16 per wave. Mean-normalization applied to the
// accumulator between phases (diag-scale commutes with right-multiplication),
// so the load->cvt->wmma path carries no multiplies.
__global__ void gemm_fused_wmma(const float* __restrict__ Asum, const float* __restrict__ H,
                                const float* __restrict__ invA,
                                const __bf16* __restrict__ WlT, const __bf16* __restrict__ WrT,
                                const float* __restrict__ bias, float* __restrict__ out,
                                int M, int relu) {
  const int lane = threadIdx.x & 31;
  const int wave = threadIdx.x >> 5;
  const int row0 = blockIdx.x * 16;
  const int col0 = wave * 16;
  const int mrow = row0 + (lane & 15);
  const int m    = (mrow < M) ? mrow : (M - 1);   // clamp (rows >= M never stored)
  const int koff = (lane < 16) ? 0 : 8;           // K-half owned by this lane group
  const int ncol = col0 + (lane & 15);
  const int rbase = row0 + ((lane < 16) ? 0 : 8); // C/D: VGPR r -> row rbase+r

  v8f acc = {0.f, 0.f, 0.f, 0.f, 0.f, 0.f, 0.f, 0.f};

  // ---- Phase 0: acc = Asum @ Wl (unnormalized sums) ----
  #pragma unroll
  for (int kb = 0; kb < DFEAT; kb += 32) {
    v16bf a, b;
    const float* arow = Asum + (long)m * DFEAT + kb + koff;
    #pragma unroll
    for (int e2 = 0; e2 < 8; ++e2) {
      a[e2]     = (__bf16)arow[e2];
      a[8 + e2] = (__bf16)arow[16 + e2];
    }
    const __bf16* wrow = WlT + (long)ncol * DFEAT + kb + koff;
    #pragma unroll
    for (int e2 = 0; e2 < 8; ++e2) {
      b[e2]     = wrow[e2];
      b[8 + e2] = wrow[16 + e2];
    }
    acc = __builtin_amdgcn_wmma_f32_16x16x32_bf16(
        false, a, false, b, (short)0, acc, false, false);
  }

  // ---- Row-scale the accumulator: diag(inv) * (Asum @ Wl) ----
  #pragma unroll
  for (int r = 0; r < 8; ++r) {
    int rr = rbase + r;
    acc[r] *= invA[(rr < M) ? rr : (M - 1)];
  }

  // ---- Phase 1: acc += H @ Wr ----
  #pragma unroll
  for (int kb = 0; kb < DFEAT; kb += 32) {
    v16bf a, b;
    const float* arow = H + (long)m * DFEAT + kb + koff;
    #pragma unroll
    for (int e2 = 0; e2 < 8; ++e2) {
      a[e2]     = (__bf16)arow[e2];
      a[8 + e2] = (__bf16)arow[16 + e2];
    }
    const __bf16* wrow = WrT + (long)ncol * DFEAT + kb + koff;
    #pragma unroll
    for (int e2 = 0; e2 < 8; ++e2) {
      b[e2]     = wrow[e2];
      b[8 + e2] = wrow[16 + e2];
    }
    acc = __builtin_amdgcn_wmma_f32_16x16x32_bf16(
        false, a, false, b, (short)0, acc, false, false);
  }

  // ---- Epilogue: bias (+ReLU), store ----
  const float bv = bias[ncol];
  #pragma unroll
  for (int r = 0; r < 8; ++r) {
    int rr = rbase + r;
    if (rr < M) {
      float v = acc[r] + bv;
      if (relu) v = fmaxf(v, 0.0f);
      out[(long)rr * DFEAT + ncol] = v;
    }
  }
}

extern "C" void kernel_launch(void* const* d_in, const int* in_sizes, int n_in,
                              void* d_out, int out_size, void* d_ws, size_t ws_size,
                              hipStream_t stream) {
  const float* x_person  = (const float*)d_in[0];
  const float* x_product = (const float*)d_in[1];
  const int*   src_pv    = (const int*)d_in[2];
  const int*   dst_pv    = (const int*)d_in[3];
  const int*   src_vp    = (const int*)d_in[4];
  const int*   dst_vp    = (const int*)d_in[5];
  const float *Wl_pv0 = (const float*)d_in[6],  *Wr_pv0 = (const float*)d_in[7],  *b_pv0 = (const float*)d_in[8];
  const float *Wl_vp0 = (const float*)d_in[9],  *Wr_vp0 = (const float*)d_in[10], *b_vp0 = (const float*)d_in[11];
  const float *Wl_pv1 = (const float*)d_in[12], *Wr_pv1 = (const float*)d_in[13], *b_pv1 = (const float*)d_in[14];
  const float *Wl_vp1 = (const float*)d_in[15], *Wr_vp1 = (const float*)d_in[16], *b_vp1 = (const float*)d_in[17];

  const int NP = in_sizes[0] / DFEAT;   // 100000
  const int NQ = in_sizes[1] / DFEAT;   // 50000
  const int E  = in_sizes[2];           // 600000

  // Workspace (floats then bf16 tail):
  //   inv_p[NP] inv_q[NQ] sum_q[NQ*128] sum_p[NP*128] h_p1[NP*128] h_q1[NQ*128]
  //   wt[8][128*128] bf16  (~154.5 MB total)
  float* ws    = (float*)d_ws;
  float* inv_p = ws;
  float* inv_q = inv_p + NP;
  float* sum_q = inv_q + NQ;
  float* sum_p = sum_q + (long)NQ * DFEAT;
  float* h_p1  = sum_p + (long)NP * DFEAT;
  float* h_q1  = h_p1  + (long)NP * DFEAT;
  __bf16* wt   = (__bf16*)(h_q1 + (long)NQ * DFEAT);
  const long WSZ = (long)DFEAT * DFEAT;
  __bf16 *WlT_pv0 = wt + 0 * WSZ, *WrT_pv0 = wt + 1 * WSZ;
  __bf16 *WlT_vp0 = wt + 2 * WSZ, *WrT_vp0 = wt + 3 * WSZ;
  __bf16 *WlT_pv1 = wt + 4 * WSZ, *WrT_pv1 = wt + 5 * WSZ;
  __bf16 *WlT_vp1 = wt + 6 * WSZ, *WrT_vp1 = wt + 7 * WSZ;

  float* out_p = (float*)d_out;
  float* out_q = out_p + (long)NP * DFEAT;

  const int TB = 256;
  auto blk = [](long n, int tb) { return (unsigned)((n + tb - 1) / tb); };

  const long nq_e = (long)NQ * DFEAT;
  const long np_e = (long)NP * DFEAT;
  const long scat = (long)E * 32;

  // Pre-transpose + bf16-convert all weight matrices (256 KB, L2-resident).
  const unsigned wblk = blk(WSZ, TB);
  transpose_w_bf16<<<wblk, TB, 0, stream>>>(Wl_pv0, WlT_pv0);
  transpose_w_bf16<<<wblk, TB, 0, stream>>>(Wr_pv0, WrT_pv0);
  transpose_w_bf16<<<wblk, TB, 0, stream>>>(Wl_vp0, WlT_vp0);
  transpose_w_bf16<<<wblk, TB, 0, stream>>>(Wr_vp0, WrT_vp0);
  transpose_w_bf16<<<wblk, TB, 0, stream>>>(Wl_pv1, WlT_pv1);
  transpose_w_bf16<<<wblk, TB, 0, stream>>>(Wr_pv1, WrT_pv1);
  transpose_w_bf16<<<wblk, TB, 0, stream>>>(Wl_vp1, WlT_vp1);
  transpose_w_bf16<<<wblk, TB, 0, stream>>>(Wr_vp1, WrT_vp1);

  // Degree counts (static across layers), inverted in place.
  fill_zero<<<blk(NP, TB), TB, 0, stream>>>(inv_p, NP);
  fill_zero<<<blk(NQ, TB), TB, 0, stream>>>(inv_q, NQ);
  count_edges<<<blk(E, TB), TB, 0, stream>>>(dst_pv, inv_q, E);
  count_edges<<<blk(E, TB), TB, 0, stream>>>(dst_vp, inv_p, E);
  invert_counts<<<blk(NQ, TB), TB, 0, stream>>>(inv_q, NQ);
  invert_counts<<<blk(NP, TB), TB, 0, stream>>>(inv_p, NP);

  // ---------------- Layer 0 ----------------
  fill_zero<<<blk(nq_e, TB), TB, 0, stream>>>(sum_q, nq_e);
  fill_zero<<<blk(np_e, TB), TB, 0, stream>>>(sum_p, np_e);
  scatter_add<<<blk(scat, TB), TB, 0, stream>>>(x_person,  src_pv, dst_pv, sum_q, E);
  scatter_add<<<blk(scat, TB), TB, 0, stream>>>(x_product, src_vp, dst_vp, sum_p, E);
  gemm_fused_wmma<<<(NQ + 15) / 16, TB, 0, stream>>>(sum_q, x_product, inv_q,
                                                     WlT_pv0, WrT_pv0, b_pv0, h_q1, NQ, 1);
  gemm_fused_wmma<<<(NP + 15) / 16, TB, 0, stream>>>(sum_p, x_person,  inv_p,
                                                     WlT_vp0, WrT_vp0, b_vp0, h_p1, NP, 1);

  // ---------------- Layer 1 ----------------
  fill_zero<<<blk(nq_e, TB), TB, 0, stream>>>(sum_q, nq_e);
  fill_zero<<<blk(np_e, TB), TB, 0, stream>>>(sum_p, np_e);
  scatter_add<<<blk(scat, TB), TB, 0, stream>>>(h_p1, src_pv, dst_pv, sum_q, E);
  scatter_add<<<blk(scat, TB), TB, 0, stream>>>(h_q1, src_vp, dst_vp, sum_p, E);
  gemm_fused_wmma<<<(NQ + 15) / 16, TB, 0, stream>>>(sum_q, h_q1, inv_q,
                                                     WlT_pv1, WrT_pv1, b_pv1, out_q, NQ, 0);
  gemm_fused_wmma<<<(NP + 15) / 16, TB, 0, stream>>>(sum_p, h_p1, inv_p,
                                                     WlT_vp1, WrT_vp1, b_vp1, out_p, NP, 0);
}
